// Encoder_57226144252448
// MI455X (gfx1250) — matrix operand order repaired and checked by reading, not compiled, not run on hardware
//
#include <hip/hip_runtime.h>

typedef __attribute__((ext_vector_type(16))) __bf16 v16bf;
typedef __attribute__((ext_vector_type(8)))  float  v8f;
typedef unsigned short u16;

#define B_   64
#define S_   256
#define U_   1024
#define H3_  3072

// ---- helpers -------------------------------------------------------------

static __device__ __forceinline__ u16 f2bf(float f) {
  unsigned u; __builtin_memcpy(&u, &f, 4);
  u = (u + 0x7FFFu + ((u >> 16) & 1u)) >> 16;   // round-to-nearest-even
  return (u16)u;
}

// B operand: lane holds column N=L%16, K={klo..klo+7}, {klo+16..klo+23} (stride H3_)
static __device__ __forceinline__ v16bf ld_b16(const u16* p) {
  v16bf v; u16* d = reinterpret_cast<u16*>(&v);
  #pragma unroll
  for (int s = 0; s < 8; ++s) { d[s] = p[(long long)s * H3_]; d[8 + s] = p[(long long)(16 + s) * H3_]; }
  return v;
}
// A operand: lane holds two 8-bf16 chunks (K={klo..klo+7}, {klo+16..klo+23})
static __device__ __forceinline__ v16bf ld_a8x2(const u16* plo, const u16* phi) {
  v16bf v;
  __builtin_memcpy(&v, plo, 16);
  __builtin_memcpy(reinterpret_cast<char*>(&v) + 16, phi, 16);
  return v;
}

static __device__ __forceinline__ float sigmoidf_(float x) {
  return 1.0f / (1.0f + __expf(-x));
}

#define WMMA_BF16(A, Bm, C) \
  __builtin_amdgcn_wmma_f32_16x16x32_bf16(false, (A), false, (Bm), (short)0, (C), false, false)

// ---- kernel 1: embedding gather -> bf16 ---------------------------------

__global__ void k_embed(const int* __restrict__ x, const float* __restrict__ emb,
                        u16* __restrict__ xe) {
  const int row = blockIdx.x;                    // 0 .. B*S-1
  const int tok = x[row];
  const float* src = emb + (long long)tok * U_;
  u16* dst = xe + (long long)row * U_;
  for (int c = threadIdx.x; c < U_; c += blockDim.x)
    dst[c] = f2bf(src[c]);
}

// ---- kernel 1b: fp32 -> bf16 weight conversion --------------------------

__global__ void k_cvt(const float* __restrict__ src, u16* __restrict__ dst, int n) {
  int i = blockIdx.x * blockDim.x + threadIdx.x;
  const int st = gridDim.x * blockDim.x;
  for (; i < n; i += st) dst[i] = f2bf(src[i]);
}

// ---- kernel 2: input projection xp = x_emb @ W + b_i (both directions) ---
// grid: (H3/128, B*S/64, 2), block 256 (8 waves). Each wave: 16x64 of C.
// Two-stage ping-pong pipeline (rolled, peeled tail): forces loads a full
// stage ahead of their WMMA; rotation movs co-execute with XDL WMMA.

__global__ void __launch_bounds__(256)
k_inproj(const u16* __restrict__ xe,
         const u16* __restrict__ wk_f, const u16* __restrict__ wk_b,
         const float* __restrict__ bias_f, const float* __restrict__ bias_b,
         float* __restrict__ xp_f, float* __restrict__ xp_b) {
  const int dir = blockIdx.z;
  const u16*   wk = dir ? wk_b : wk_f;
  const float* bi = dir ? bias_b : bias_f;   // row 0 of (2,3072)
  float*       xp = dir ? xp_b : xp_f;

  const int wave = threadIdx.x >> 5;
  const int lane = threadIdx.x & 31;
  const int m_base = blockIdx.y * 64 + (wave >> 1) * 16;
  const int n_base = blockIdx.x * 128 + (wave & 1) * 64;

  v8f acc[4];
  #pragma unroll
  for (int i = 0; i < 4; ++i)
    #pragma unroll
    for (int r = 0; r < 8; ++r) acc[i][r] = 0.0f;

  const int arow = m_base + (lane & 15);
  const int klo  = (lane < 16) ? 0 : 8;
  const u16* Abase = xe + (long long)arow * U_ + klo;
  const u16* Bbase = wk + (long long)klo * H3_ + n_base + (lane & 15);   // N col = lane%16

  v16bf a0, a1, b0[4], b1[4];

  // prologue: set0 <- k-block 0
  a0 = ld_a8x2(Abase, Abase + 16);
  #pragma unroll
  for (int i = 0; i < 4; ++i) b0[i] = ld_b16(Bbase + i * 16);

  #pragma unroll 1
  for (int kk2 = 0; kk2 < 15; ++kk2) {
    const int kB = kk2 * 64 + 32;
    const int kA = kk2 * 64 + 64;
    {
      const u16* Bn = Bbase + (long long)kB * H3_;
      a1 = ld_a8x2(Abase + kB, Abase + kB + 16);
      #pragma unroll
      for (int i = 0; i < 4; ++i) b1[i] = ld_b16(Bn + i * 16);
    }
    #pragma unroll
    for (int i = 0; i < 4; ++i) acc[i] = WMMA_BF16(a0, b0[i], acc[i]);
    {
      const u16* Bn = Bbase + (long long)kA * H3_;
      a0 = ld_a8x2(Abase + kA, Abase + kA + 16);
      #pragma unroll
      for (int i = 0; i < 4; ++i) b0[i] = ld_b16(Bn + i * 16);
    }
    #pragma unroll
    for (int i = 0; i < 4; ++i) acc[i] = WMMA_BF16(a1, b1[i], acc[i]);
  }
  // tail: set0 holds k=960; load set1 at k=992
  {
    const int kB = 992;
    const u16* Bn = Bbase + (long long)kB * H3_;
    a1 = ld_a8x2(Abase + kB, Abase + kB + 16);
    #pragma unroll
    for (int i = 0; i < 4; ++i) b1[i] = ld_b16(Bn + i * 16);
  }
  #pragma unroll
  for (int i = 0; i < 4; ++i) acc[i] = WMMA_BF16(a0, b0[i], acc[i]);
  #pragma unroll
  for (int i = 0; i < 4; ++i) acc[i] = WMMA_BF16(a1, b1[i], acc[i]);

  const int mh = (lane >> 4) << 3;
  const int n_lane = lane & 15;
  #pragma unroll
  for (int i = 0; i < 4; ++i) {
    const int col = n_base + i * 16 + n_lane;
    const float bv = bi[col];
    #pragma unroll
    for (int r = 0; r < 8; ++r) {
      const int m = m_base + mh + r;
      xp[(long long)m * H3_ + col] = acc[i][r] + bv;
    }
  }
}

// ---- kernel 3: persistent GRU scan --------------------------------------
// grid: (4 batch-tiles, 2 directions), block 512 (16 waves, 4/SIMD for
// latency hiding on the sequential critical path). Each WG owns 16 batch
// rows for the full scan; each wave covers 4 N-tiles of 16 cols.
// LDS: double-buffered h (fp32 64KB x2) + bf16 A copy (32KB x2) = 192KB.
// Same two-stage ping-pong K pipeline (A from LDS, 3 gate-B tiles from L2).

__global__ void __launch_bounds__(512)
k_recur(const int* __restrict__ x,
        const u16* __restrict__ wr_f, const u16* __restrict__ wr_b,
        const float* __restrict__ bias_f, const float* __restrict__ bias_b,
        const float* __restrict__ xp_f, const float* __restrict__ xp_b,
        float* __restrict__ out_f, float* __restrict__ out_b) {
  extern __shared__ char smem[];
  float* hf0 = (float*)smem;                 // [16][1024] fp32
  float* hf1 = hf0 + 16 * U_;
  u16*   hb0 = (u16*)(hf1 + 16 * U_);        // [16][1024] bf16
  u16*   hb1 = hb0 + 16 * U_;

  const int dir = blockIdx.y;
  const int b0_ = blockIdx.x * 16;
  const u16*   wr = dir ? wr_b : wr_f;
  const float* br = (dir ? bias_b : bias_f) + H3_;  // row 1 = recurrent bias
  const float* xp = dir ? xp_b : xp_f;
  float* outp = dir ? out_b : out_f;

  for (int i = threadIdx.x; i < 16 * U_; i += blockDim.x) {
    hf0[i] = 0.0f; hf1[i] = 0.0f; hb0[i] = 0; hb1[i] = 0;
  }
  __syncthreads();

  const int wave = threadIdx.x >> 5;       // 0..15
  const int lane = threadIdx.x & 31;
  const int arow = lane & 15;
  const int klo  = (lane < 16) ? 0 : 8;
  const int mh   = (lane >> 4) << 3;
  const int n_lane = lane & 15;

  int p = 0;
  for (int t = 0; t < S_; ++t) {
    const int pos = dir ? (S_ - 1 - t) : t;
    const float* hf_old = p ? hf1 : hf0;
    float*       hf_new = p ? hf0 : hf1;
    const u16*   hb_old = p ? hb1 : hb0;
    u16*         hb_new = p ? hb0 : hb1;

    #pragma unroll 1
    for (int i = 0; i < 4; ++i) {           // 4 N-tiles of 16 cols per wave
      const int col0 = (wave * 4 + i) * 16;
      v8f az, ar, ah;
      #pragma unroll
      for (int r = 0; r < 8; ++r) { az[r] = 0.f; ar[r] = 0.f; ah[r] = 0.f; }

      const u16* Abase = hb_old + arow * U_ + klo;
      const u16* Bbase = wr + (long long)klo * H3_ + col0 + (lane & 15);  // N col = lane%16

      v16bf a0, a1, z0, g0, h0, z1, g1, h1;

      // prologue: set0 <- k-block 0
      a0 = ld_a8x2(Abase, Abase + 16);
      z0 = ld_b16(Bbase);
      g0 = ld_b16(Bbase + U_);
      h0 = ld_b16(Bbase + 2 * U_);

      #pragma unroll 1
      for (int kk2 = 0; kk2 < 15; ++kk2) {
        const int kB = kk2 * 64 + 32;
        const int kA = kk2 * 64 + 64;
        {
          const u16* Bn = Bbase + (long long)kB * H3_;
          __builtin_prefetch(Bn + 64 * (long long)H3_, 0, 0);  // first-pass HBM->L2
          a1 = ld_a8x2(Abase + kB, Abase + kB + 16);
          z1 = ld_b16(Bn);
          g1 = ld_b16(Bn + U_);
          h1 = ld_b16(Bn + 2 * U_);
        }
        az = WMMA_BF16(a0, z0, az);
        ar = WMMA_BF16(a0, g0, ar);
        ah = WMMA_BF16(a0, h0, ah);
        {
          const u16* Bn = Bbase + (long long)kA * H3_;
          a0 = ld_a8x2(Abase + kA, Abase + kA + 16);
          z0 = ld_b16(Bn);
          g0 = ld_b16(Bn + U_);
          h0 = ld_b16(Bn + 2 * U_);
        }
        az = WMMA_BF16(a1, z1, az);
        ar = WMMA_BF16(a1, g1, ar);
        ah = WMMA_BF16(a1, h1, ah);
      }
      // tail: set0 holds k=960; load set1 at k=992
      {
        const int kB = 992;
        const u16* Bn = Bbase + (long long)kB * H3_;
        a1 = ld_a8x2(Abase + kB, Abase + kB + 16);
        z1 = ld_b16(Bn);
        g1 = ld_b16(Bn + U_);
        h1 = ld_b16(Bn + 2 * U_);
      }
      az = WMMA_BF16(a0, z0, az);
      ar = WMMA_BF16(a0, g0, ar);
      ah = WMMA_BF16(a0, h0, ah);
      az = WMMA_BF16(a1, z1, az);
      ar = WMMA_BF16(a1, g1, ar);
      ah = WMMA_BF16(a1, h1, ah);

      // Gate math: same (VGPR r, lane) holds z/r/h-tilde for element (m,n).
      #pragma unroll
      for (int r = 0; r < 8; ++r) {
        const int m = mh + r;
        const int b = b0_ + m;
        const int col = col0 + n_lane;
        const long long xi = (long long)(b * S_ + pos) * H3_ + col;
        const float xz = xp[xi];
        const float xr = xp[xi + U_];
        const float xh = xp[xi + 2 * U_];
        const float rz = az[r] + br[col];
        const float rr = ar[r] + br[U_ + col];
        const float rh = ah[r] + br[2 * U_ + col];
        const float z  = sigmoidf_(xz + rz);
        const float rg = sigmoidf_(xr + rr);
        const float hh = tanhf(xh + rg * rh);
        const float hold = hf_old[m * U_ + col];
        float hnew = z * hold + (1.0f - z) * hh;
        if (x[b * S_ + pos] == 0) hnew = hold;   // mask: keep previous h
        hf_new[m * U_ + col] = hnew;
        hb_new[m * U_ + col] = f2bf(hnew);
        outp[(long long)(b * S_ + pos) * U_ + col] = hnew;
      }
    }
    __syncthreads();   // one barrier per timestep (double-buffered h)
    p ^= 1;
  }
}

// ---- kernel 4: out = out_f + out_b --------------------------------------

__global__ void k_add(float* __restrict__ out, const float* __restrict__ add, int n) {
  int i = blockIdx.x * blockDim.x + threadIdx.x;
  const int st = gridDim.x * blockDim.x;
  for (; i < n; i += st) out[i] += add[i];
}

// ---- host glue -----------------------------------------------------------

extern "C" void kernel_launch(void* const* d_in, const int* in_sizes, int n_in,
                              void* d_out, int out_size, void* d_ws, size_t ws_size,
                              hipStream_t stream) {
  (void)in_sizes; (void)n_in; (void)out_size; (void)ws_size;
  const int*   x   = (const int*)d_in[0];
  const float* emb = (const float*)d_in[1];
  const float* kf  = (const float*)d_in[2];
  const float* rf  = (const float*)d_in[3];
  const float* bf  = (const float*)d_in[4];
  const float* kb  = (const float*)d_in[5];
  const float* rb  = (const float*)d_in[6];
  const float* bb  = (const float*)d_in[7];
  float* out = (float*)d_out;

  char* ws = (char*)d_ws;
  size_t off = 0;
  auto alloc = [&](size_t bytes) -> char* {
    char* pp = ws + off;
    off += (bytes + 255) & ~(size_t)255;
    return pp;
  };
  u16*   xe   = (u16*)  alloc((size_t)B_ * S_ * U_ * 2);
  u16*   wkf  = (u16*)  alloc((size_t)U_ * H3_ * 2);
  u16*   wkb  = (u16*)  alloc((size_t)U_ * H3_ * 2);
  u16*   wrf  = (u16*)  alloc((size_t)U_ * H3_ * 2);
  u16*   wrb  = (u16*)  alloc((size_t)U_ * H3_ * 2);
  float* xpf  = (float*)alloc((size_t)B_ * S_ * H3_ * 4);
  float* xpb  = (float*)alloc((size_t)B_ * S_ * H3_ * 4);
  float* outb = (float*)alloc((size_t)B_ * S_ * U_ * 4);

  k_embed<<<B_ * S_, 256, 0, stream>>>(x, emb, xe);
  const int wn = U_ * H3_;
  k_cvt<<<2048, 256, 0, stream>>>(kf, wkf, wn);
  k_cvt<<<2048, 256, 0, stream>>>(kb, wkb, wn);
  k_cvt<<<2048, 256, 0, stream>>>(rf, wrf, wn);
  k_cvt<<<2048, 256, 0, stream>>>(rb, wrb, wn);

  dim3 g2(H3_ / 128, (B_ * S_) / 64, 2);
  k_inproj<<<g2, 256, 0, stream>>>(xe, wkf, wkb, bf, bb, xpf, xpb);

  (void)hipFuncSetAttribute((const void*)k_recur,
                            hipFuncAttributeMaxDynamicSharedMemorySize,
                            192 * 1024);
  dim3 g3(4, 2);
  k_recur<<<g3, 512, 192 * 1024, stream>>>(x, wrf, wrb, bf, bb, xpf, xpb,
                                           out, outb);

  k_add<<<4096, 256, 0, stream>>>(out, outb, B_ * S_ * U_);
}
